// TemNonLocalBlock_781684048673
// MI455X (gfx1250) — compile-verified, matrix operand order checked by reading, not run on hardware
//
#include <hip/hip_runtime.h>

typedef __attribute__((ext_vector_type(16))) __bf16 v16bf;
typedef __attribute__((ext_vector_type(8)))  float  v8f;

union Frag16 { v16bf v; unsigned int u[8]; };

__device__ __forceinline__ __bf16 tobf(float f) { return (__bf16)f; }

// Load a 16x32 bf16 WMMA fragment (A: row=M, or B: row=N) from an LDS matrix
// stored row-major with K contiguous. Per ISA 7.12.2 16-bit layout:
// lane 0-15 -> row, lane 16-31 -> same row, K offset +8;
// VGPR v holds K pair (v>>2)*16 + half*8 + (v&3)*2.
__device__ __forceinline__ v16bf load_frag(const __bf16* __restrict__ base,
                                           int row0, int ldk, int k0, int lane) {
  const int r     = row0 + (lane & 15);
  const int half4 = (lane >> 4) * 4;
  const unsigned int* p = reinterpret_cast<const unsigned int*>(base + r * ldk + k0);
  Frag16 f;
#pragma unroll
  for (int v = 0; v < 8; ++v)
    f.u[v] = p[(v >> 2) * 8 + half4 + (v & 3)];
  return f.v;
}

__global__ __launch_bounds__(128, 1)
void tnl_fused_kernel(const float* __restrict__ x,
                      const float* __restrict__ w_g,  const float* __restrict__ b_g,
                      const float* __restrict__ w_th, const float* __restrict__ b_th,
                      const float* __restrict__ w_ph, const float* __restrict__ b_ph,
                      const float* __restrict__ proj,
                      const float* __restrict__ w_out, const float* __restrict__ b_out,
                      float* __restrict__ out) {
  constexpr int C = 64, T = 2048, J = 64, CI = 32, NCAT = 96;

  __shared__ __align__(16) float  xs[C * J];        // x slice f32, [c][j] (residual)
  __shared__ __align__(16) __bf16 xT[J * C];        // x^T bf16,  [j][c]  (GEMM1 A)
  __shared__ __align__(16) __bf16 WcatT[NCAT * C];  // [n][c], n = g|th|ph (GEMM1 B)
  __shared__ __align__(16) float  P1s[J * CI];      // proj[:, :Ci] f32 (row-dot)
  __shared__ __align__(16) __bf16 P2s[J * CI];      // proj[:, Ci:] bf16 (GEMM2 A)
  __shared__ __align__(16) __bf16 Wouts[C * CI];    // w_out [c][o] bf16 (GEMM4 B)
  __shared__ float  bcat[NCAT];
  __shared__ float  bouts[C];
  __shared__ __align__(16) __bf16 ths[J * CI];      // theta [j][o]
  __shared__ __align__(16) __bf16 phs[J * CI];      // phi   [i][o]   (GEMM2 B)
  __shared__ __align__(16) __bf16 gTs[CI * J];      // g^T   [o][i]   (GEMM3 B)
  __shared__ float  a_s[J];
  __shared__ __align__(16) __bf16 fs[J * J];        // relu(a+bb)/J   (GEMM3 A)
  __shared__ __align__(16) __bf16 ys[J * CI];       // y [j][o]       (GEMM4 A)
  __shared__ __align__(16) float  outs[C * J];      // staged output [c][j]

  const int tid   = threadIdx.x;
  const int lane  = tid & 31;
  const int wave  = tid >> 5;
  const int l16   = lane & 15;
  const int halfm = (lane >> 4) * 8;

  const int t = blockIdx.x & (T - 1);
  const int b = blockIdx.x >> 11;                    // / T
  const size_t xbase    = (size_t)b * C * T * J + (size_t)t * J;
  const size_t crstride = (size_t)T * J;

  // Prefetch the x slice rows (global_prefetch_b8) while weights stream in.
  for (int c = wave; c < C; c += 4)
    __builtin_prefetch(&x[xbase + (size_t)c * crstride + (size_t)(lane & 15) * 4], 0, 3);

  // ---- stage weights (tiny; L2-resident across blocks) ----
  for (int idx = tid; idx < NCAT * C; idx += 128) {
    int n = idx >> 6, c = idx & 63;
    float w = (n < 32) ? w_g[n * C + c]
            : (n < 64) ? w_th[(n - 32) * C + c]
                       : w_ph[(n - 64) * C + c];
    WcatT[idx] = tobf(w);
  }
  for (int idx = tid; idx < J * CI; idx += 128) {
    int j = idx >> 5, c = idx & 31;
    P1s[idx]   = proj[j * (2 * CI) + c];
    P2s[idx]   = tobf(proj[j * (2 * CI) + CI + c]);
    Wouts[idx] = tobf(w_out[idx]);                   // (C,Ci) row-major == [c][o]
  }
  if (tid < NCAT)
    bcat[tid] = (tid < 32) ? b_g[tid] : (tid < 64) ? b_th[tid - 32] : b_ph[tid - 64];
  if (tid < C) bouts[tid] = b_out[tid];

  // ---- stage x slice: f32 [c][j] + transposed bf16 [j][c] ----
  for (int q = tid; q < C * (J / 4); q += 128) {     // 1024 float4
    int c = q >> 4, j4 = q & 15;
    float4 v = *reinterpret_cast<const float4*>(&x[xbase + (size_t)c * crstride + j4 * 4]);
    reinterpret_cast<float4*>(xs)[q] = v;
    int j = j4 * 4;
    xT[(j + 0) * C + c] = tobf(v.x);
    xT[(j + 1) * C + c] = tobf(v.y);
    xT[(j + 2) * C + c] = tobf(v.z);
    xT[(j + 3) * C + c] = tobf(v.w);
  }
  __syncthreads();

  // ---- GEMM1: [g|th|ph](j,o) = x^T(j,c) * Wcat^T(c,n) + bias.  24 tiles, K=64.
  {
    const int m0 = wave * 16;
    v16bf a0 = load_frag(xT, m0, C, 0, lane);
    v16bf a1 = load_frag(xT, m0, C, 32, lane);
#pragma unroll
    for (int nt = 0; nt < 6; ++nt) {
      v16bf b0 = load_frag(WcatT, nt * 16, C, 0, lane);
      v16bf b1 = load_frag(WcatT, nt * 16, C, 32, lane);
      v8f d = {};
      d = __builtin_amdgcn_wmma_f32_16x16x32_bf16(false, a0, false, b0, (short)0, d, false, false);
      d = __builtin_amdgcn_wmma_f32_16x16x32_bf16(false, a1, false, b1, (short)0, d, false, false);
      const float bias = bcat[nt * 16 + l16];
#pragma unroll
      for (int v = 0; v < 8; ++v) {
        int m = m0 + halfm + v;                      // j (or i)
        float val = d[v] + bias;
        if (nt < 2) {                                // g -> transposed [o][i]
          gTs[(nt * 16 + l16) * J + m] = tobf(val);
        } else if (nt < 4) {                         // theta [j][o]
          ths[m * CI + (nt - 2) * 16 + l16] = tobf(val);
        } else {                                     // phi   [i][o]
          phs[m * CI + (nt - 4) * 16 + l16] = tobf(val);
        }
      }
    }
  }
  __syncthreads();

  // ---- GEMM2: bb(j,i) = P2(j,c) * phi^T(c,i).  16 tiles, K=32.  Plus row-dot a(j).
  v8f bbreg[4];
  {
    const int m0 = wave * 16;
    v16bf a2 = load_frag(P2s, m0, CI, 0, lane);
#pragma unroll
    for (int it = 0; it < 4; ++it) {
      v16bf b2 = load_frag(phs, it * 16, CI, 0, lane);
      v8f d = {};
      d = __builtin_amdgcn_wmma_f32_16x16x32_bf16(false, a2, false, b2, (short)0, d, false, false);
      bbreg[it] = d;
    }
  }
  if (tid < J) {
    float acc = 0.f;
#pragma unroll
    for (int c = 0; c < CI; ++c)
      acc += (float)ths[tid * CI + c] * P1s[tid * CI + c];
    a_s[tid] = acc;
  }
  __syncthreads();

  // ---- f = relu(a + bb) / J,  stored bf16 [j][i]
  {
    const int m0 = wave * 16;
    const float inv = 1.0f / (float)J;
#pragma unroll
    for (int it = 0; it < 4; ++it) {
#pragma unroll
      for (int v = 0; v < 8; ++v) {
        int m = m0 + halfm + v;
        float val = bbreg[it][v] + a_s[m];
        fs[m * J + it * 16 + l16] = tobf(fmaxf(val, 0.f) * inv);
      }
    }
  }
  __syncthreads();

  // ---- GEMM3: y(j,o) = f(j,i) * g(i,o).  8 tiles, K=64.
  {
    const int m0 = wave * 16;
    v16bf a0 = load_frag(fs, m0, J, 0, lane);
    v16bf a1 = load_frag(fs, m0, J, 32, lane);
#pragma unroll
    for (int ot = 0; ot < 2; ++ot) {
      v16bf b0 = load_frag(gTs, ot * 16, J, 0, lane);
      v16bf b1 = load_frag(gTs, ot * 16, J, 32, lane);
      v8f d = {};
      d = __builtin_amdgcn_wmma_f32_16x16x32_bf16(false, a0, false, b0, (short)0, d, false, false);
      d = __builtin_amdgcn_wmma_f32_16x16x32_bf16(false, a1, false, b1, (short)0, d, false, false);
#pragma unroll
      for (int v = 0; v < 8; ++v)
        ys[(m0 + halfm + v) * CI + ot * 16 + l16] = tobf(d[v]);
    }
  }
  __syncthreads();

  // ---- GEMM4: out(j,c) = y(j,o) * w_out^T(o,c) + b_out.  16 tiles, K=32.
  {
    const int m0 = wave * 16;
    v16bf a4 = load_frag(ys, m0, CI, 0, lane);
#pragma unroll
    for (int ct = 0; ct < 4; ++ct) {
      v16bf b4 = load_frag(Wouts, ct * 16, CI, 0, lane);
      v8f d = {};
      d = __builtin_amdgcn_wmma_f32_16x16x32_bf16(false, a4, false, b4, (short)0, d, false, false);
      const int cc   = ct * 16 + l16;
      const float bc = bouts[cc];
#pragma unroll
      for (int v = 0; v < 8; ++v)
        outs[cc * J + (m0 + halfm + v)] = d[v] + bc;   // staged [c][j]
    }
  }
  __syncthreads();

  // ---- residual add + coalesced float4 store, same layout as x ----
  for (int q = tid; q < C * (J / 4); q += 128) {
    int c = q >> 4, j4 = q & 15;
    float4 v = reinterpret_cast<const float4*>(outs)[q];
    float4 r = reinterpret_cast<const float4*>(xs)[q];
    v.x += r.x; v.y += r.y; v.z += r.z; v.w += r.w;
    *reinterpret_cast<float4*>(&out[xbase + (size_t)c * crstride + j4 * 4]) = v;
  }
}

extern "C" void kernel_launch(void* const* d_in, const int* in_sizes, int n_in,
                              void* d_out, int out_size, void* d_ws, size_t ws_size,
                              hipStream_t stream) {
  (void)in_sizes; (void)n_in; (void)out_size; (void)d_ws; (void)ws_size;
  const float* x     = (const float*)d_in[0];
  const float* w_g   = (const float*)d_in[1];
  const float* b_g   = (const float*)d_in[2];
  const float* w_th  = (const float*)d_in[3];
  const float* b_th  = (const float*)d_in[4];
  const float* w_ph  = (const float*)d_in[5];
  const float* b_ph  = (const float*)d_in[6];
  const float* proj  = (const float*)d_in[7];
  const float* w_out = (const float*)d_in[8];
  const float* b_out = (const float*)d_in[9];
  float* out = (float*)d_out;

  const int B = 8, T = 2048;
  dim3 grid(B * T), block(128);
  tnl_fused_kernel<<<grid, block, 0, stream>>>(x, w_g, b_g, w_th, b_th, w_ph, b_ph,
                                               proj, w_out, b_out, out);
}
